// MLP_21741124452873
// MI455X (gfx1250) — matrix operand order repaired and checked by reading, not compiled, not run on hardware
//
#include <hip/hip_runtime.h>

typedef __bf16 bf16;
typedef __attribute__((ext_vector_type(8)))  bf16  bf16x8;
typedef __attribute__((ext_vector_type(16))) bf16  bf16x16;
typedef __attribute__((ext_vector_type(8)))  float f32x8;

#define E_DIM 8
#define T_DIM 2048
#define H_DIM 1024
#define F_DIM 4096
#define FC    128   // F chunk per outer iteration
#define KC1   128   // K sub-chunk for stage-1 w1 staging (double-buffered)
#define KC2   32    // K sub-chunk for stage-2 w2 staging (double-buffered)
#define MT    32    // tokens per workgroup (fast path)

__device__ __forceinline__ float gelu_tanh(float v) {
    const float c = 0.7978845608028654f;
    float u = c * (v + 0.044715f * v * v * v);
    return 0.5f * v * (1.0f + tanhf(u));
}

__device__ __forceinline__ bf16x16 cat8(bf16x8 lo, bf16x8 hi) {
    return __builtin_shufflevector(lo, hi, 0,1,2,3,4,5,6,7,8,9,10,11,12,13,14,15);
}

// CDNA5 async global->LDS copy, 16B per lane, tracked by ASYNCcnt.
// GV addressing: VDST = LDS byte offset (vgpr), VADDR = 64-bit global address.
__device__ __forceinline__ void async_b128(unsigned lds_off, const void* gaddr) {
    asm volatile("global_load_async_to_lds_b128 %0, %1, off"
                 :: "v"(lds_off), "v"((unsigned long long)gaddr)
                 : "memory");
}

__device__ __forceinline__ void wait_async0() {
#if __has_builtin(__builtin_amdgcn_s_wait_asynccnt)
    __builtin_amdgcn_s_wait_asynccnt(0);
#else
    asm volatile("s_wait_asynccnt 0x0" ::: "memory");
#endif
}

// ---------------------------------------------------------------------------
// Pre-pass: transpose + fp32->bf16 convert, src[e][R][C] -> dst[e][C][R].
// ---------------------------------------------------------------------------
__global__ __launch_bounds__(256) void transpose_cvt_bf16(
        const float* __restrict__ src, bf16* __restrict__ dst, int R, int C) {
    __shared__ float tile[32][33];
    const int e  = blockIdx.z;
    const float* s = src + (long)e * R * C;
    bf16*        d = dst + (long)e * R * C;
    const int c0 = blockIdx.x * 32;
    const int r0 = blockIdx.y * 32;
    const int tx = threadIdx.x;      // 0..31
    const int ty = threadIdx.y;      // 0..7
    #pragma unroll
    for (int j = 0; j < 32; j += 8)
        tile[ty + j][tx] = s[(long)(r0 + ty + j) * C + (c0 + tx)];
    __syncthreads();
    #pragma unroll
    for (int j = 0; j < 32; j += 8)
        d[(long)(c0 + ty + j) * R + (r0 + tx)] = (bf16)tile[tx][ty + j];
}

// ---------------------------------------------------------------------------
// Fast path: fused MLP on pre-transposed bf16 weights, async double-buffered
// LDS staging. Grid (T/32, E), block 512 = 16 waves (wave32).
// w1t layout [F][H]; w2t layout [H][F] (tiles contiguous along K).
// ---------------------------------------------------------------------------
__global__ __launch_bounds__(512) void mlp_fused_wmma_bf16(
        const float* __restrict__ x,
        const bf16*  __restrict__ w1t,
        const bf16*  __restrict__ w2t,
        float*       __restrict__ out) {
    extern __shared__ char smem[];
    bf16* xs  = (bf16*)smem;                      // [MT][H]       64 KB
    bf16* ws1[2]; ws1[0] = xs + MT * H_DIM;       // [FC][KC1] x2  64 KB
    ws1[1] = ws1[0] + FC * KC1;
    bf16* ws2[2]; ws2[0] = ws1[1] + FC * KC1;     // [H][KC2]  x2 128 KB
    ws2[1] = ws2[0] + H_DIM * KC2;
    bf16* hts = ws2[1] + H_DIM * KC2;             // [MT][FC]       8 KB

    const unsigned sbase = __builtin_amdgcn_groupstaticsize();
    unsigned ws1_off[2], ws2_off[2];
    ws1_off[0] = sbase + (unsigned)((char*)ws1[0] - smem);
    ws1_off[1] = sbase + (unsigned)((char*)ws1[1] - smem);
    ws2_off[0] = sbase + (unsigned)((char*)ws2[0] - smem);
    ws2_off[1] = sbase + (unsigned)((char*)ws2[1] - smem);

    const int tid  = threadIdx.x;
    const int wv   = tid >> 5;
    const int lane = tid & 31;
    const int nl   = lane & 15;
    const int half = lane >> 4;
    const int mt   = wv >> 3;            // 0..1
    const int nw   = wv & 7;             // 0..7

    const int e  = blockIdx.y;
    const int t0 = blockIdx.x * MT;

    const float* xe  = x   + ((long)e * T_DIM + t0) * H_DIM;
    const bf16*  w1e = w1t + (long)e * F_DIM * H_DIM;
    const bf16*  w2e = w2t + (long)e * H_DIM * F_DIM;
    float*       oe  = out + ((long)e * T_DIM + t0) * H_DIM;

    // Issue async copies of one w1 tile: cols f0..f0+FC, k in [kc, kc+KC1).
    auto stage1_issue = [&](int b, int f0, int kc) {
        #pragma unroll
        for (int j = 0; j < (FC * KC1) / 8 / 512; ++j) {     // 4 per thread
            int u = tid + j * 512;
            int c = u >> 4;                  // KC1/8 == 16 units per column
            int koff = (u & 15) * 8;
            async_b128(ws1_off[b] + (unsigned)(c * KC1 + koff) * 2,
                       w1e + (long)(f0 + c) * H_DIM + kc + koff);
        }
    };
    // Issue async copies of one w2 tile: n = 0..H, k in [f0+kc2, +KC2).
    auto stage2_issue = [&](int b, int f0, int kc2) {
        #pragma unroll
        for (int j = 0; j < (H_DIM * KC2) / 8 / 512; ++j) {  // 8 per thread
            int u = tid + j * 512;
            int n = u >> 2;                  // KC2/8 == 4 units per row
            int koff = (u & 3) * 8;
            async_b128(ws2_off[b] + (unsigned)(n * KC2 + koff) * 2,
                       w2e + (long)n * F_DIM + f0 + kc2 + koff);
        }
    };

    // Stage the MT x H x-tile once (fp32 -> bf16), coalesced.
    for (int i = tid; i < MT * H_DIM; i += 512) xs[i] = (bf16)xe[i];

    const f32x8 fz = {0.f, 0.f, 0.f, 0.f, 0.f, 0.f, 0.f, 0.f};
    f32x8 acc[8];
    #pragma unroll
    for (int i = 0; i < 8; ++i) acc[i] = fz;

    for (int f0 = 0; f0 < F_DIM; f0 += FC) {
        // -------- stage 1: htile[MT][FC] = gelu(x @ w1[:, f0:f0+FC]) --------
        f32x8 c1 = fz;
        stage1_issue(0, f0, 0);
        wait_async0();
        __syncthreads();                       // also publishes xs on 1st chunk
        int pb = 0;
        for (int kc = 0; kc < H_DIM; kc += KC1) {
            if (kc + KC1 < H_DIM) stage1_issue(pb ^ 1, f0, kc + KC1);
            const bf16* arow = xs + (mt * 16 + nl) * H_DIM + kc;   // A row
            const bf16* bcol = ws1[pb] + (nw * 16 + nl) * KC1;     // B col
            #pragma unroll
            for (int kk = 0; kk < KC1; kk += 32) {
                bf16x16 A = cat8(*(const bf16x8*)(arow + kk + 8 * half),
                                 *(const bf16x8*)(arow + kk + 16 + 8 * half));
                bf16x16 B = cat8(*(const bf16x8*)(bcol + kk + 8 * half),
                                 *(const bf16x8*)(bcol + kk + 16 + 8 * half));
                c1 = __builtin_amdgcn_wmma_f32_16x16x32_bf16(
                        false, A, false, B, (short)0, c1, false, false);
            }
            wait_async0();                     // my next-buffer copies done
            __syncthreads();                   // everyone's done; buf readable
            pb ^= 1;
        }
        // GELU epilogue; C layout: VGPR r -> M = r + 8*half, N = nl.
        #pragma unroll
        for (int r = 0; r < 8; ++r)
            hts[(mt * 16 + r + 8 * half) * FC + nw * 16 + nl] =
                (bf16)gelu_tanh(c1[r]);

        // -------- stage 2: acc += htile @ w2[f0:f0+FC, :] -------------------
        stage2_issue(0, f0, 0);
        wait_async0();
        __syncthreads();                       // also publishes hts
        pb = 0;
        for (int kc2 = 0; kc2 < FC; kc2 += KC2) {
            if (kc2 + KC2 < FC) stage2_issue(pb ^ 1, f0, kc2 + KC2);
            const bf16* arow2 = hts + (mt * 16 + nl) * FC + kc2;
            bf16x16 A = cat8(*(const bf16x8*)(arow2 + 8 * half),
                             *(const bf16x8*)(arow2 + 16 + 8 * half));
            #pragma unroll
            for (int nt = 0; nt < 8; ++nt) {
                const bf16* bcol2 = ws2[pb] + (nw * 128 + nt * 16 + nl) * KC2;
                bf16x16 B = cat8(*(const bf16x8*)(bcol2 + 8 * half),
                                 *(const bf16x8*)(bcol2 + 16 + 8 * half));
                acc[nt] = __builtin_amdgcn_wmma_f32_16x16x32_bf16(
                        false, A, false, B, (short)0, acc[nt], false, false);
            }
            wait_async0();
            __syncthreads();
            pb ^= 1;
        }
    }

    // -------- epilogue: out[MT, H] fp32 -------------------------------------
    #pragma unroll
    for (int nt = 0; nt < 8; ++nt) {
        #pragma unroll
        for (int r = 0; r < 8; ++r) {
            int m = mt * 16 + r + 8 * half;
            oe[(long)m * H_DIM + nw * 128 + nt * 16 + nl] = acc[nt][r];
        }
    }
}

// ---------------------------------------------------------------------------
// Fallback (sync staging from fp32 weights) if d_ws is too small.
// Grid (T/16, E), block 256 = 8 waves.
// ---------------------------------------------------------------------------
__global__ __launch_bounds__(256) void mlp_fused_wmma_f32src(
        const float* __restrict__ x,
        const float* __restrict__ w1,
        const float* __restrict__ w2,
        float* __restrict__ out) {
    extern __shared__ char smem[];
    bf16* xs   = (bf16*)smem;            // [16][H]
    bf16* ws1t = xs   + 16 * H_DIM;      // [FC][KC1]
    bf16* ws2t = ws1t + FC * KC1;        // [H][KC2]
    bf16* hts  = ws2t + H_DIM * KC2;     // [16][FC]

    const int tid  = threadIdx.x;
    const int wv   = tid >> 5;
    const int lane = tid & 31;
    const int nl   = lane & 15;
    const int half = lane >> 4;

    const int e  = blockIdx.y;
    const int t0 = blockIdx.x * 16;

    const float* xe  = x  + ((long)e * T_DIM + t0) * H_DIM;
    const float* w1e = w1 + (long)e * H_DIM * F_DIM;
    const float* w2e = w2 + (long)e * F_DIM * H_DIM;
    float*       oe  = out + ((long)e * T_DIM + t0) * H_DIM;

    for (int i = tid; i < 16 * H_DIM; i += 256) xs[i] = (bf16)xe[i];

    const f32x8 fz = {0.f, 0.f, 0.f, 0.f, 0.f, 0.f, 0.f, 0.f};
    f32x8 acc[8];
    #pragma unroll
    for (int i = 0; i < 8; ++i) acc[i] = fz;

    for (int f0 = 0; f0 < F_DIM; f0 += FC) {
        f32x8 c1 = fz;
        for (int kc = 0; kc < H_DIM; kc += KC1) {
            __syncthreads();
            for (int i = tid; i < KC1 * FC; i += 256) {
                int r = i >> 7, c = i & (FC - 1);
                ws1t[c * KC1 + r] = (bf16)w1e[(long)(kc + r) * F_DIM + (f0 + c)];
            }
            __syncthreads();
            const bf16* arow = xs + nl * H_DIM + kc;
            const bf16* bcol = ws1t + (wv * 16 + nl) * KC1;
            #pragma unroll
            for (int kk = 0; kk < KC1; kk += 32) {
                bf16x16 A = cat8(*(const bf16x8*)(arow + kk + 8 * half),
                                 *(const bf16x8*)(arow + kk + 16 + 8 * half));
                bf16x16 B = cat8(*(const bf16x8*)(bcol + kk + 8 * half),
                                 *(const bf16x8*)(bcol + kk + 16 + 8 * half));
                c1 = __builtin_amdgcn_wmma_f32_16x16x32_bf16(
                        false, A, false, B, (short)0, c1, false, false);
            }
        }
        #pragma unroll
        for (int r = 0; r < 8; ++r)
            hts[(r + 8 * half) * FC + wv * 16 + nl] = (bf16)gelu_tanh(c1[r]);

        for (int kc2 = 0; kc2 < FC; kc2 += KC2) {
            __syncthreads();
            for (int i = tid; i < KC2 * H_DIM; i += 256) {
                int r = i >> 10, c = i & (H_DIM - 1);
                ws2t[c * KC2 + r] = (bf16)w2e[(long)(f0 + kc2 + r) * H_DIM + c];
            }
            __syncthreads();
            const bf16* arow2 = hts + nl * FC + kc2;
            bf16x16 A = cat8(*(const bf16x8*)(arow2 + 8 * half),
                             *(const bf16x8*)(arow2 + 16 + 8 * half));
            #pragma unroll
            for (int nt = 0; nt < 8; ++nt) {
                const bf16* bcol2 = ws2t + (wv * 128 + nt * 16 + nl) * KC2;
                bf16x16 B = cat8(*(const bf16x8*)(bcol2 + 8 * half),
                                 *(const bf16x8*)(bcol2 + 16 + 8 * half));
                acc[nt] = __builtin_amdgcn_wmma_f32_16x16x32_bf16(
                        false, A, false, B, (short)0, acc[nt], false, false);
            }
        }
    }
    #pragma unroll
    for (int nt = 0; nt < 8; ++nt)
        #pragma unroll
        for (int r = 0; r < 8; ++r)
            oe[(long)(r + 8 * half) * H_DIM + wv * 128 + nt * 16 + nl] = acc[nt][r];
}

extern "C" void kernel_launch(void* const* d_in, const int* in_sizes, int n_in,
                              void* d_out, int out_size, void* d_ws, size_t ws_size,
                              hipStream_t stream) {
    (void)in_sizes; (void)n_in; (void)out_size;
    const float* x  = (const float*)d_in[0];
    const float* w1 = (const float*)d_in[1];
    const float* w2 = (const float*)d_in[2];
    float* out = (float*)d_out;

    const size_t wElems   = (size_t)E_DIM * F_DIM * H_DIM;        // per weight
    const size_t wsNeeded = 2 * wElems * sizeof(bf16);            // 128 MiB

    if (d_ws != nullptr && ws_size >= wsNeeded) {
        bf16* w1t = (bf16*)d_ws;           // [E][F][H]
        bf16* w2t = w1t + wElems;          // [E][H][F]

        dim3 tb(32, 8);
        transpose_cvt_bf16<<<dim3(F_DIM / 32, H_DIM / 32, E_DIM), tb, 0, stream>>>(
            w1, w1t, H_DIM, F_DIM);   // [H][F] -> [F][H]
        transpose_cvt_bf16<<<dim3(H_DIM / 32, F_DIM / 32, E_DIM), tb, 0, stream>>>(
            w2, w2t, F_DIM, H_DIM);   // [F][H] -> [H][F]

        // LDS: xs 64K + 2x ws1 64K + 2x ws2 128K + hts 8K = 264 KB.
        const size_t lds_bytes =
            (size_t)(MT * H_DIM + 2 * FC * KC1 + 2 * H_DIM * KC2 + MT * FC)
            * sizeof(bf16);
        mlp_fused_wmma_bf16<<<dim3(T_DIM / MT, E_DIM), dim3(512),
                              lds_bytes, stream>>>(x, w1t, w2t, out);
    } else {
        const size_t lds_bytes =
            (size_t)(16 * H_DIM + FC * KC1 + H_DIM * KC2 + 16 * FC) * sizeof(bf16);
        mlp_fused_wmma_f32src<<<dim3(T_DIM / 16, E_DIM), dim3(256),
                                lds_bytes, stream>>>(x, w1, w2, out);
    }
}
